// EdgeLevelAttentionLayer_65910568124775
// MI455X (gfx1250) — matrix operand order repaired and checked by reading, not compiled, not run on hardware
//
#include <hip/hip_runtime.h>
#include <hip/hip_bf16.h>

// ---------------------------------------------------------------------------
// Problem constants (from the reference)
// ---------------------------------------------------------------------------
#define N_NODES   4096
#define N_EDGES   8192
#define NODE_FEAT 64
#define EDGE_FEAT 32
#define OUT_COLS  128   // 64 (node path) + 64 (edge path)
#define CAP       64    // max neighbors kept per row (deg mean ~16, binomial; 64 is >>10 sigma)

// ---------------------------------------------------------------------------
// Workspace layout (in 4-byte elements)
// ---------------------------------------------------------------------------
constexpr long OFF_U1      = 0;        // 32   W_edge @ p_node[:64]
constexpr long OFF_U2      = 32;       // 64   W_node @ p_node[64:]
constexpr long OFF_U3      = 96;       // 32   W_edge @ p_edge[:64]
constexpr long OFF_U4      = 128;      // 32   W_edge @ p_edge[64:]
constexpr long OFF_SSELF_N = 160;      // 8192
constexpr long OFF_SNB_N   = 8352;     // 4096
constexpr long OFF_SSELF_E = 12448;    // 8192
constexpr long OFF_SNB_E   = 20640;    // 8192
constexpr long OFF_MEANS_N = 28832;    // 8192*64
constexpr long OFF_MEANS_E = 553120;   // 8192*32
constexpr long OFF_CNT_N   = 815264;   // 8192 ints
constexpr long OFF_CNT_E   = 823456;   // 8192 ints (contiguous with CNT_N)
constexpr long OFF_NBR_N   = 831648;   // 8192*CAP ints
constexpr long OFF_NBR_E   = 1355936;  // 8192*CAP ints
// total = 1,880,224 elems = ~7.52 MB

__device__ __forceinline__ float lrelu(float x) { return x > 0.0f ? x : 0.01f * x; }

typedef __attribute__((ext_vector_type(4)))  float  f4v;
typedef __attribute__((ext_vector_type(16))) __bf16 v16bf;
typedef __attribute__((ext_vector_type(8)))  float  v8f;

// ---------------------------------------------------------------------------
// 0) zero the neighbor counters (d_ws is poisoned; must be reset every call)
// ---------------------------------------------------------------------------
__global__ void zero_cnt_kernel(int* __restrict__ cnt, int n) {
    int i = blockIdx.x * blockDim.x + threadIdx.x;
    if (i < n) cnt[i] = 0;
}

// ---------------------------------------------------------------------------
// 1) u-vectors: fold W @ p-slices into four tiny vectors (one block, 64 thr)
// ---------------------------------------------------------------------------
__global__ void compute_u_kernel(const float* __restrict__ Wn,   // [64,64]
                                 const float* __restrict__ We,   // [32,64]
                                 const float* __restrict__ pe,   // [128]
                                 const float* __restrict__ pn,   // [128]
                                 float* __restrict__ ws) {
    int t = threadIdx.x;  // 0..63
    if (t < 32) {
        float a = 0.f, b = 0.f, c = 0.f;
        #pragma unroll 8
        for (int j = 0; j < 64; ++j) {
            float w = We[t * 64 + j];
            a += w * pn[j];
            b += w * pe[j];
            c += w * pe[64 + j];
        }
        ws[OFF_U1 + t] = a;
        ws[OFF_U3 + t] = b;
        ws[OFF_U4 + t] = c;
    }
    float d = 0.f;
    #pragma unroll 8
    for (int j = 0; j < 64; ++j) d += Wn[t * 64 + j] * pn[64 + j];
    ws[OFF_U2 + t] = d;
}

// ---------------------------------------------------------------------------
// 2) per-edge / per-node attention scores (dot against u-vectors)
// ---------------------------------------------------------------------------
__global__ void compute_s_kernel(const float* __restrict__ ef,  // [8192,32]
                                 const float* __restrict__ nf,  // [4096,64]
                                 float* __restrict__ ws) {
    int id = blockIdx.x * blockDim.x + threadIdx.x;
    if (id < N_EDGES) {
        const float* e = ef + (long)id * EDGE_FEAT;
        float a = 0.f, b = 0.f, c = 0.f;
        #pragma unroll 8
        for (int j = 0; j < EDGE_FEAT; ++j) {
            float v = e[j];
            a += v * ws[OFF_U1 + j];
            b += v * ws[OFF_U3 + j];
            c += v * ws[OFF_U4 + j];
        }
        ws[OFF_SSELF_N + id] = a;   // s_self for node-level attention
        ws[OFF_SSELF_E + id] = b;   // s_self for edge-level attention
        ws[OFF_SNB_E   + id] = c;   // s_nb   for edge-level attention
    } else if (id < N_EDGES + N_NODES) {
        int n = id - N_EDGES;
        const float* p = nf + (long)n * NODE_FEAT;
        float d = 0.f;
        #pragma unroll 8
        for (int j = 0; j < NODE_FEAT; ++j) d += p[j] * ws[OFF_U2 + j];
        ws[OFF_SNB_N + n] = d;
    }
}

// ---------------------------------------------------------------------------
// 3) stream a dense [rows, 8192] binary matrix once (non-temporal float4),
//    scatter nonzeros (r,c) into per-OUTPUT-row (= column c) neighbor lists.
//    (Both A_n and A_e in the reference are transposes of the stored inputs.)
// ---------------------------------------------------------------------------
__global__ void scan_adj_kernel(const float* __restrict__ adj, long total4,
                                int* __restrict__ cnt, int* __restrict__ nbrs) {
    const long stride = (long)gridDim.x * blockDim.x;
    const f4v* p = (const f4v*)adj;
    for (long i = (long)blockIdx.x * blockDim.x + threadIdx.x; i < total4; i += stride) {
        f4v v = __builtin_nontemporal_load(p + i);   // 384 MB streamed, no reuse
        if (v.x != 0.f || v.y != 0.f || v.z != 0.f || v.w != 0.f) {
            long lin   = i << 2;
            int  r     = (int)(lin >> 13);      // cols == 8192
            int  cbase = (int)(lin & 8191);
            float vals[4] = {v.x, v.y, v.z, v.w};
            #pragma unroll
            for (int j = 0; j < 4; ++j) {
                if (vals[j] != 0.f) {
                    int c   = cbase + j;
                    int pos = atomicAdd(&cnt[c], 1);
                    if (pos < CAP) nbrs[(long)c * CAP + pos] = r;
                }
            }
        }
    }
}

// ---------------------------------------------------------------------------
// 4) exact softmax over sparse support + aggregation into means.
//    One 64-thread block per output row. Neighbor list is sorted so the
//    float summation order is deterministic across replays.
// ---------------------------------------------------------------------------
template <int FEAT>
__global__ void agg_kernel(const int*   __restrict__ cnt,
                           const int*   __restrict__ nbrs,
                           const float* __restrict__ s_self_arr,
                           const float* __restrict__ s_nb_arr,
                           const float* __restrict__ feats,     // [ncols? no: neighbor rows, FEAT]
                           float*       __restrict__ means,     // [8192, FEAT]
                           int ncols) {
    __shared__ int   nb_s[CAP];
    __shared__ float w[CAP];
    __shared__ float red[64];

    const int e = blockIdx.x;
    const int t = threadIdx.x;  // 64 threads

    int c = cnt[e];
    if (c > CAP) c = CAP;
    const float s_self = s_self_arr[e];
    const float base   = lrelu(s_self);

    if (t < c) nb_s[t] = nbrs[(long)e * CAP + t];
    __syncthreads();
    if (t == 0) {  // insertion sort (c ~ 8..16) for deterministic fp order
        for (int i = 1; i < c; ++i) {
            int key = nb_s[i], j = i - 1;
            while (j >= 0 && nb_s[j] > key) { nb_s[j + 1] = nb_s[j]; --j; }
            nb_s[j + 1] = key;
        }
    }
    __syncthreads();

    float term = -3.4e38f;
    if (t < c) term = lrelu(s_self + s_nb_arr[nb_s[t]]);

    red[t] = term;
    __syncthreads();
    for (int off = 32; off > 0; off >>= 1) {
        if (t < off) red[t] = fmaxf(red[t], red[t + off]);
        __syncthreads();
    }
    const float m = fmaxf(red[0], base);
    __syncthreads();

    float wv = 0.f;
    if (t < c) { wv = expf(term - m); w[t] = wv; }
    red[t] = wv;
    __syncthreads();
    for (int off = 32; off > 0; off >>= 1) {
        if (t < off) red[t] += red[t + off];
        __syncthreads();
    }
    const float Z     = red[0] + (float)(ncols - c) * expf(base - m);
    const float scale = 1.f / (Z * (float)ncols);

    if (t < FEAT) {
        float acc = 0.f;
        for (int i = 0; i < c; ++i)
            acc += w[i] * feats[(long)nb_s[i] * FEAT + t];
        means[(long)e * FEAT + t] = acc * scale;
    }
}

// ---------------------------------------------------------------------------
// 5) out = lrelu(means @ W) via v_wmma_f32_16x16x32_bf16, fused into the
//    concatenated [8192,128] output. One wave per 16x16 tile; EXEC all-ones.
// ---------------------------------------------------------------------------
template <int K>
__global__ void gemm_lrelu_wmma_kernel(const float* __restrict__ A,  // [8192, K]
                                       const float* __restrict__ B,  // [K, 64]
                                       float* __restrict__ out,      // [8192, 128]
                                       int colOff) {
    const int tile = blockIdx.x;
    const int tn   = tile & 3;     // 64/16 = 4 N-tiles
    const int tm   = tile >> 2;
    const int lane = threadIdx.x;  // wave32, all lanes active
    const int r    = lane & 15;
    const int h    = lane >> 4;

    v8f c = {};
    #pragma unroll
    for (int kt = 0; kt < K / 32; ++kt) {
        v16bf a, b;
        #pragma unroll
        for (int i = 0; i < 16; ++i) {
            // A frag (16x32, 16-bit): elem i of lane (r,h) holds K = 16*(i/8) + i%8 + 8h
            int ka = 16 * (i >> 3) + (i & 7) + 8 * h;
            a[i] = (__bf16)A[(long)(tm * 16 + r) * K + kt * 32 + ka];
            // B frag (32x16): elem i of lane (r,h) holds K = 16*(i/8) + i%8 + 8h; col = r
            int kb = 16 * (i >> 3) + (i & 7) + 8 * h;
            b[i] = (__bf16)B[(long)(kt * 32 + kb) * 64 + tn * 16 + r];
        }
        c = __builtin_amdgcn_wmma_f32_16x16x32_bf16(
                /*neg_a=*/false, a, /*neg_b=*/false, b,
                /*c_mod=*/(short)0, c, /*reuse_a=*/false, /*reuse_b=*/false);
    }

    #pragma unroll
    for (int j = 0; j < 8; ++j) {
        // D frag: VGPR j holds M = j + 8h, N = lane%16
        int row = tm * 16 + j + 8 * h;
        int col = tn * 16 + r;
        float v = c[j];
        out[(long)row * OUT_COLS + colOff + col] = lrelu(v);
    }
}

// ---------------------------------------------------------------------------
// launch
// ---------------------------------------------------------------------------
extern "C" void kernel_launch(void* const* d_in, const int* in_sizes, int n_in,
                              void* d_out, int out_size, void* d_ws, size_t ws_size,
                              hipStream_t stream) {
    const float* node_features = (const float*)d_in[0];  // [4096,64]
    const float* edge_features = (const float*)d_in[1];  // [8192,32]
    const float* edge_to_edge  = (const float*)d_in[2];  // [8192,8192]
    const float* node_to_edge  = (const float*)d_in[3];  // [4096,8192]
    const float* W_node        = (const float*)d_in[4];  // [64,64]
    const float* W_edge        = (const float*)d_in[5];  // [32,64]
    const float* p_edge        = (const float*)d_in[6];  // [128]
    const float* p_node        = (const float*)d_in[7];  // [128]
    float* out = (float*)d_out;                          // [8192,128]
    float* ws  = (float*)d_ws;
    int*   wsi = (int*)d_ws;

    // 0) reset neighbor counters (CNT_N and CNT_E are contiguous)
    zero_cnt_kernel<<<64, 256, 0, stream>>>(wsi + OFF_CNT_N, 2 * N_EDGES);

    // 1) fold projection vectors
    compute_u_kernel<<<1, 64, 0, stream>>>(W_node, W_edge, p_edge, p_node, ws);

    // 2) per-row attention scores
    compute_s_kernel<<<(N_EDGES + N_NODES) / 256, 256, 0, stream>>>(
        edge_features, node_features, ws);

    // 3) stream both adjacency matrices once (the roofline-dominant 384 MB)
    scan_adj_kernel<<<2048, 256, 0, stream>>>(
        node_to_edge, (long)N_NODES * N_EDGES / 4,
        wsi + OFF_CNT_N, wsi + OFF_NBR_N);
    scan_adj_kernel<<<4096, 256, 0, stream>>>(
        edge_to_edge, (long)N_EDGES * N_EDGES / 4,
        wsi + OFF_CNT_E, wsi + OFF_NBR_E);

    // 4) sparse softmax + aggregation
    agg_kernel<NODE_FEAT><<<N_EDGES, 64, 0, stream>>>(
        wsi + OFF_CNT_N, wsi + OFF_NBR_N,
        ws + OFF_SSELF_N, ws + OFF_SNB_N,
        node_features, ws + OFF_MEANS_N, N_NODES);
    agg_kernel<EDGE_FEAT><<<N_EDGES, 64, 0, stream>>>(
        wsi + OFF_CNT_E, wsi + OFF_NBR_E,
        ws + OFF_SSELF_E, ws + OFF_SNB_E,
        edge_features, ws + OFF_MEANS_E, N_EDGES);

    // 5) final dense projections via WMMA, fused lrelu + concat store
    gemm_lrelu_wmma_kernel<NODE_FEAT><<<(N_EDGES / 16) * 4, 32, 0, stream>>>(
        ws + OFF_MEANS_N, W_node, out, /*colOff=*/0);
    gemm_lrelu_wmma_kernel<EDGE_FEAT><<<(N_EDGES / 16) * 4, 32, 0, stream>>>(
        ws + OFF_MEANS_E, W_edge, out, /*colOff=*/64);
}
